// Attention_76493367542415
// MI455X (gfx1250) — matrix operand order repaired and checked by reading, not compiled, not run on hardware
//
// CDNA5 / gfx1250 Transformer-XL rel-pos attention (B=8,T=1024,H=8,DK=64).
// All heavy math via v_wmma_f32_16x16x32_bf16 (fp32 accumulate).
// Attention tiles staged to LDS by the Tensor Data Mover (tensor_load_to_lds,
// TENSORcnt). Pipeline: [fuse weights] -> [Q/K/V + pos projections, bf16] ->
// [flash attention w/ banded rel-shift] -> [fused output GEMM].
#include <hip/hip_runtime.h>
#include <hip/hip_bf16.h>

constexpr int kB = 8, kT = 1024, kH = 8, kDK = 64, kInner = 512, kP = 2047;

typedef __attribute__((ext_vector_type(16))) __bf16 v16bf;
typedef __attribute__((ext_vector_type(8)))  __bf16 v8bf;
typedef __attribute__((ext_vector_type(8)))  float  v8f;
typedef __attribute__((ext_vector_type(4)))  unsigned int u32x4;
typedef __attribute__((ext_vector_type(8)))  int i32x8;
typedef __attribute__((ext_vector_type(4)))  int i32x4;

union V16U { v16bf v; v8bf h[2]; };
union PK2 { __bf16 h[2]; unsigned int u; };

enum { MODE_PLAIN = 0, MODE_QUV, MODE_KP, MODE_VP, MODE_POS, MODE_OUT };

__device__ inline v8f wmma_bf16(v16bf a, v16bf b, v8f c) {
  return __builtin_amdgcn_wmma_f32_16x16x32_bf16(false, a, false, b, (short)0, c,
                                                 false, false);
}

__device__ inline v8f vzero() {
  v8f z;
#pragma unroll
  for (int i = 0; i < 8; ++i) z[i] = 0.0f;
  return z;
}

// cross-lane reductions within a 16-lane half-wave (C-frag rows are per-VGPR,
// columns per-lane; xor masks 1,2,4,8 stay inside the half)
__device__ inline float rowmax16(float x) {
  x = fmaxf(x, __shfl_xor(x, 1, 32));
  x = fmaxf(x, __shfl_xor(x, 2, 32));
  x = fmaxf(x, __shfl_xor(x, 4, 32));
  x = fmaxf(x, __shfl_xor(x, 8, 32));
  return x;
}
__device__ inline float rowsum16(float x) {
  x += __shfl_xor(x, 1, 32);
  x += __shfl_xor(x, 2, 32);
  x += __shfl_xor(x, 4, 32);
  x += __shfl_xor(x, 8, 32);
  return x;
}

// ---- Tensor Data Mover: 2D tile (data_size = 2B) global -> LDS ------------
// D# per CDNA5 ISA ch.8: group0 = {count/lds_addr/global_addr/type},
// group1 = {data_size, tensor dims (set huge: no OOB), tile dims, stride}.
// This toolchain exposes the 6-arg builtin (extra int32x8 group, zero-filled).
__device__ inline void tdm_load_tile(unsigned lds_off, const void* gptr,
                                     unsigned tile0, unsigned tile1,
                                     unsigned stride0) {
  unsigned long long ga = (unsigned long long)(size_t)gptr;
  u32x4 g0;
  g0[0] = 1u;                                                  // count=1 (valid)
  g0[1] = lds_off;                                             // LDS byte addr
  g0[2] = (unsigned)ga;                                        // gaddr[31:0]
  g0[3] = (unsigned)((ga >> 32) & 0x1FFFFFFull) | 0x80000000u; // gaddr[56:32]|type=2
  i32x8 g1;
  g1[0] = 1 << 16;                            // data_size code 1 = 2 bytes
  g1[1] = 0;                                  // no atomic barrier; dim0 lo16=0
  g1[2] = 0x4000;                             // tensor_dim0 = 1<<30 (hi16)
  g1[3] = (int)(0x4000u | (tile0 << 16));     // tensor_dim1 = 1<<30; tile_dim0
  g1[4] = (int)tile1;                         // tile_dim1 (tile_dim2 = 0)
  g1[5] = (int)stride0;                       // tensor_dim0_stride[31:0]
  g1[6] = 0;
  g1[7] = 0;
  i32x4 gz4;
  gz4[0] = 0; gz4[1] = 0; gz4[2] = 0; gz4[3] = 0;
  i32x8 gz8;
#pragma unroll
  for (int i = 0; i < 8; ++i) gz8[i] = 0;
  __builtin_amdgcn_tensor_load_to_lds(g0, g1, gz4, gz4, gz8, 0);
}

struct GemmArgs {
  const float*  Af;  const __bf16* Ab;  int lda;
  const float*  Bf;  const __bf16* Bb;  int ldb;
  int M, N, K;
  float scale;
  const float* bias; const float* bias_u; const float* bias_v;
  __bf16* out0; __bf16* out1; float* outf;
};

// Generic bf16-WMMA GEMM: 128 thr (4 waves), 64x64 tile, K step 32.
// Dtypes and epilogue are compile-time; B is staged transposed into LDS with
// packed (2 x bf16 -> b32) stores so the stage loop is branch-free and vector.
template <bool AF32, bool BF32, int MODE>
__global__ __launch_bounds__(128) void gemm_kernel(GemmArgs g) {
  __shared__ __align__(32) __bf16 Bt[64 * 32];  // [n][k]
  const int tid = threadIdx.x;
  const int wave = tid >> 5, lane = tid & 31;
  const int ln = lane & 15, hi = lane >> 4;
  const int n0 = blockIdx.x * 64;
  const int m0 = blockIdx.y * 64;
  int arow = m0 + wave * 16 + ln;
  if (arow > g.M - 1) arow = g.M - 1;
  const int ko = hi ? 16 : 0;    // A-frag K phase (contiguous k = 16h + s)
  const int kb2 = hi ? 16 : 0;   // B-frag K phase

  v8f acc[4];
#pragma unroll
  for (int j = 0; j < 4; ++j) acc[j] = vzero();

  for (int kk = 0; kk < g.K; kk += 32) {
    // stage Bt[n][k] = B[kk+k][n0+n]; two coalesced row loads -> one b32 store
#pragma unroll
    for (int it = 0; it < 8; ++it) {
      int idx = tid + it * 128;          // 1024 k-pair slots
      int n = idx & 63, k2 = (idx >> 6) * 2;
      PK2 pk;
      if constexpr (BF32) {
        pk.h[0] = (__bf16)g.Bf[(size_t)(kk + k2) * g.ldb + n0 + n];
        pk.h[1] = (__bf16)g.Bf[(size_t)(kk + k2 + 1) * g.ldb + n0 + n];
      } else {
        pk.h[0] = g.Bb[(size_t)(kk + k2) * g.ldb + n0 + n];
        pk.h[1] = g.Bb[(size_t)(kk + k2 + 1) * g.ldb + n0 + n];
      }
      *(unsigned int*)&Bt[n * 32 + k2] = pk.u;
    }
    __syncthreads();

    V16U a;
    if constexpr (AF32) {
      const float* ap = g.Af + (size_t)arow * g.lda + kk + ko;
      float4 f0 = *(const float4*)(ap);
      float4 f1 = *(const float4*)(ap + 4);
      float4 f2 = *(const float4*)(ap + 8);
      float4 f3 = *(const float4*)(ap + 12);
      a.v[0]  = (__bf16)f0.x; a.v[1]  = (__bf16)f0.y;
      a.v[2]  = (__bf16)f0.z; a.v[3]  = (__bf16)f0.w;
      a.v[4]  = (__bf16)f1.x; a.v[5]  = (__bf16)f1.y;
      a.v[6]  = (__bf16)f1.z; a.v[7]  = (__bf16)f1.w;
      a.v[8]  = (__bf16)f2.x; a.v[9]  = (__bf16)f2.y;
      a.v[10] = (__bf16)f2.z; a.v[11] = (__bf16)f2.w;
      a.v[12] = (__bf16)f3.x; a.v[13] = (__bf16)f3.y;
      a.v[14] = (__bf16)f3.z; a.v[15] = (__bf16)f3.w;
    } else {
      const __bf16* ap = g.Ab + (size_t)arow * g.lda + kk + ko;
      a.h[0] = *(const v8bf*)(ap);
      a.h[1] = *(const v8bf*)(ap + 8);
    }

#pragma unroll
    for (int j = 0; j < 4; ++j) {
      v16bf b = *(const v16bf*)&Bt[(16 * j + ln) * 32 + kb2];
      acc[j] = wmma_bf16(a.v, b, acc[j]);
    }
    __syncthreads();
  }

#pragma unroll
  for (int j = 0; j < 4; ++j) {
#pragma unroll
    for (int v = 0; v < 8; ++v) {
      int r = m0 + wave * 16 + v + 8 * hi;
      if (r >= g.M) continue;
      int c = n0 + 16 * j + ln;
      float val = acc[j][v];
      if constexpr (MODE == MODE_PLAIN) {
        g.out0[(size_t)r * g.N + c] = (__bf16)val;
      } else if constexpr (MODE == MODE_QUV) {  // Qu & Qv, fold bias + 1/sqrt(DK)
        int b = r / kT, t = r % kT, h = c >> 6, d = c & 63;
        size_t o = ((size_t)(b * kH + h) * kT + t) * kDK + d;
        float q = val + g.bias[c];
        g.out0[o] = (__bf16)((q + g.bias_u[c]) * g.scale);
        g.out1[o] = (__bf16)((q + g.bias_v[c]) * g.scale);
      } else if constexpr (MODE == MODE_KP) {
        int b = r / kT, t = r % kT, h = c >> 6, d = c & 63;
        g.out0[((size_t)(b * kH + h) * kT + t) * kDK + d] = (__bf16)(val + g.bias[c]);
      } else if constexpr (MODE == MODE_VP) {  // V stored (b,h,dk,t) for PV B-frags
        int b = r / kT, t = r % kT, h = c >> 6, d = c & 63;
        g.out0[((size_t)(b * kH + h) * kDK + d) * kT + t] = (__bf16)(val + g.bias[c]);
      } else if constexpr (MODE == MODE_POS) {  // (h, 2047, dk)
        int h = c >> 6, d = c & 63;
        g.out0[((size_t)h * kP + r) * kDK + d] = (__bf16)val;
      } else {  // MODE_OUT
        g.outf[(size_t)r * g.N + c] = val + g.bias[c];
      }
    }
  }
}

// fused output bias: b_out = b_ao @ w_o + b_o   (tiny GEMV)
__global__ void bout_kernel(const float* b_ao, const float* w_o, const float* b_o,
                            float* bout) {
  int n = threadIdx.x;
  float acc = b_o[n];
  for (int k = 0; k < kInner; ++k) acc += b_ao[k] * w_o[(size_t)k * kInner + n];
  bout[n] = acc;
}

// Flash attention with banded rel-shift BD term.
// grid (T/64, B*H), 128 thr (4 waves). Each wave: 16 query rows.
// scores[q,k] = Qu[q].K[k] + Qv[q].P[T-1-q+k]   (scale pre-folded)
// K / V^T / P tiles are DMA'd into LDS by the TDM (one descriptor per wave).
__global__ __launch_bounds__(128) void attn_kernel(
    const __bf16* __restrict__ Qu, const __bf16* __restrict__ Qv,
    const __bf16* __restrict__ Kg, const __bf16* __restrict__ Vt,
    const __bf16* __restrict__ Pg, __bf16* __restrict__ Ao) {
  __shared__ __align__(32) __bf16 Kl[64 * 64];        // [key][dk]
  __shared__ __align__(32) __bf16 Vl[64 * 64];        // [dk][key]
  __shared__ __align__(32) __bf16 Pl[128 * 64];       // [m-W0][dk]
  __shared__ __align__(32) __bf16 Prob[4][16 * 64];   // per-wave [q][key]

  const int tid = threadIdx.x;
  const int wave = tid >> 5, lane = tid & 31;
  const int ln = lane & 15, hi = lane >> 4;
  const int bh = blockIdx.y;
  const int q0 = blockIdx.x * 64;

  const __bf16* Qup = Qu + (size_t)bh * kT * kDK;
  const __bf16* Qvp = Qv + (size_t)bh * kT * kDK;
  const __bf16* Kp  = Kg + (size_t)bh * kT * kDK;
  const __bf16* Vp  = Vt + (size_t)bh * kDK * kT;
  const __bf16* Pp  = Pg + (size_t)(bh % kH) * kP * kDK;

  // LDS byte offsets for the TDM descriptors (flat addr low 32 bits = LDS addr)
  const unsigned ldsK = (unsigned)(size_t)(void*)Kl;
  const unsigned ldsV = (unsigned)(size_t)(void*)Vl;
  const unsigned ldsP = (unsigned)(size_t)(void*)Pl;
  const int wv = __builtin_amdgcn_readfirstlane(wave);  // scalar branch for TDM

  const int ko = hi ? 16 : 0;
  const int kb16 = hi ? 16 : 0;
  const int qrow = q0 + wave * 16 + ln;

  // Qu/Qv A-fragments held in registers for the whole kernel (k-dim 64 = 2 frags)
  V16U qa[2], qva[2];
#pragma unroll
  for (int f = 0; f < 2; ++f) {
    const __bf16* p0 = Qup + (size_t)qrow * kDK + 32 * f + ko;
    qa[f].h[0] = *(const v8bf*)(p0);
    qa[f].h[1] = *(const v8bf*)(p0 + 8);
    const __bf16* p1 = Qvp + (size_t)qrow * kDK + 32 * f + ko;
    qva[f].h[0] = *(const v8bf*)(p1);
    qva[f].h[1] = *(const v8bf*)(p1 + 8);
  }

  float rmax[8], rsum[8];
  v8f o[4];
#pragma unroll
  for (int v = 0; v < 8; ++v) { rmax[v] = -3.0e38f; rsum[v] = 0.0f; }
#pragma unroll
  for (int s = 0; s < 4; ++s) o[s] = vzero();

  for (int kb = 0; kb < kT; kb += 64) {
    __syncthreads();  // previous iteration done with LDS tiles
    if (wv == 0) {            // K block: contiguous 8KB, 1D tile
      tdm_load_tile(ldsK, Kp + (size_t)kb * kDK, 4096u, 1u, 4096u);
    } else if (wv == 1) {     // V^T block: 64 rows x 64 els, row stride T
      tdm_load_tile(ldsV, Vp + (size_t)kb, 64u, 64u, (unsigned)kT);
    } else if (wv == 2) {     // P window: 128 consecutive rows, contiguous 16KB
      int W0 = kT - 64 - q0 + kb;
      tdm_load_tile(ldsP, Pp + (size_t)W0 * kDK, 8192u, 1u, 8192u);
    }
    __builtin_amdgcn_s_wait_tensorcnt((short)0);
    __syncthreads();

    // AC = Qu @ K^T  (4 key subtiles, k-dim 64 -> 2 chained WMMA)
    v8f s[4];
#pragma unroll
    for (int j = 0; j < 4; ++j) {
      v8f z = vzero();
      v16bf b0 = *(const v16bf*)&Kl[(16 * j + ln) * kDK + 0 + kb16];
      v16bf b1 = *(const v16bf*)&Kl[(16 * j + ln) * kDK + 32 + kb16];
      z = wmma_bf16(qa[0].v, b0, z);
      z = wmma_bf16(qa[1].v, b1, z);
      s[j] = z;
    }

    // banded BD strip: U[16 x 80] = Qv @ P_window^T (wave window start 16-aligned)
    v8f u[5];
    const int sw = 48 - 16 * wave;
#pragma unroll
    for (int t5 = 0; t5 < 5; ++t5) {
      v8f z = vzero();
      v16bf b0 = *(const v16bf*)&Pl[(sw + 16 * t5 + ln) * kDK + 0 + kb16];
      v16bf b1 = *(const v16bf*)&Pl[(sw + 16 * t5 + ln) * kDK + 32 + kb16];
      z = wmma_bf16(qva[0].v, b0, z);
      z = wmma_bf16(qva[1].v, b1, z);
      u[t5] = z;
    }
    // diagonal realign: score(qi, 16j+ln) += U(qi, 15+16j+ln-qi)
#pragma unroll
    for (int j = 0; j < 4; ++j) {
#pragma unroll
      for (int v = 0; v < 8; ++v) {
        int qi = v + 8 * hi;
        int d = 15 + ln - qi;               // 0..30 -> frag j or j+1
        int srcl = hi * 16 + (d & 15);      // same half, same row mapping
        float lo = __shfl(u[j][v], srcl, 32);
        float hv = __shfl(u[j + 1][v], srcl, 32);
        s[j][v] += (d >= 16) ? hv : lo;
      }
    }

    // online softmax (row stats are VGPR-indexed, replicated across half-lanes)
    float corr[8];
#pragma unroll
    for (int v = 0; v < 8; ++v) {
      float bm = fmaxf(fmaxf(s[0][v], s[1][v]), fmaxf(s[2][v], s[3][v]));
      bm = rowmax16(bm);
      float nm = fmaxf(rmax[v], bm);
      corr[v] = exp2f((rmax[v] - nm) * 1.44269504f);
      rmax[v] = nm;
      float psum = 0.0f;
#pragma unroll
      for (int j = 0; j < 4; ++j) {
        float p = exp2f((s[j][v] - nm) * 1.44269504f);
        psum += p;
        Prob[wave][(v + 8 * hi) * 64 + 16 * j + ln] = (__bf16)p;
      }
      rsum[v] = rsum[v] * corr[v] + psum;  // lane-partial; reduced at the end
    }
#pragma unroll
    for (int ss = 0; ss < 4; ++ss)
#pragma unroll
      for (int v = 0; v < 8; ++v) o[ss][v] *= corr[v];
    __syncthreads();

    // PV: O += P @ V   (A from per-wave prob tile, B from V^T)
    V16U pa[2];
#pragma unroll
    for (int f = 0; f < 2; ++f) {
      const __bf16* pp = &Prob[wave][ln * 64 + 32 * f + ko];
      pa[f].h[0] = *(const v8bf*)(pp);
      pa[f].h[1] = *(const v8bf*)(pp + 8);
    }
#pragma unroll
    for (int ss = 0; ss < 4; ++ss) {
      v16bf b0 = *(const v16bf*)&Vl[(16 * ss + ln) * 64 + 0 + kb16];
      v16bf b1 = *(const v16bf*)&Vl[(16 * ss + ln) * 64 + 32 + kb16];
      o[ss] = wmma_bf16(pa[0].v, b0, o[ss]);
      o[ss] = wmma_bf16(pa[1].v, b1, o[ss]);
    }
  }

  float inv[8];
#pragma unroll
  for (int v = 0; v < 8; ++v) inv[v] = 1.0f / rowsum16(rsum[v]);

  const int b = bh / kH, h = bh % kH;
#pragma unroll
  for (int ss = 0; ss < 4; ++ss)
#pragma unroll
    for (int v = 0; v < 8; ++v) {
      int qi = v + 8 * hi;
      int r = b * kT + q0 + wave * 16 + qi;
      int c = h * kDK + 16 * ss + ln;
      Ao[(size_t)r * kInner + c] = (__bf16)(o[ss][v] * inv[v]);
    }
}

extern "C" void kernel_launch(void* const* d_in, const int* in_sizes, int n_in,
                              void* d_out, int out_size, void* d_ws, size_t ws_size,
                              hipStream_t stream) {
  (void)in_sizes; (void)n_in; (void)out_size; (void)ws_size;
  const float* x      = (const float*)d_in[0];
  const float* pos    = (const float*)d_in[1];
  const float* w_qkv  = (const float*)d_in[2];
  const float* w_q    = (const float*)d_in[3];
  const float* b_q    = (const float*)d_in[4];
  const float* w_k    = (const float*)d_in[5];
  const float* b_k    = (const float*)d_in[6];
  const float* w_v    = (const float*)d_in[7];
  const float* b_v    = (const float*)d_in[8];
  const float* w_pos  = (const float*)d_in[9];
  const float* bias_u = (const float*)d_in[10];
  const float* bias_v = (const float*)d_in[11];
  const float* w_ao   = (const float*)d_in[12];
  const float* b_ao   = (const float*)d_in[13];
  const float* w_o    = (const float*)d_in[14];
  const float* b_o    = (const float*)d_in[15];

  char* ws = (char*)d_ws;
  size_t cur = 0;
  auto alloc = [&](size_t bytes) {
    size_t off = cur; cur += (bytes + 255) & ~(size_t)255; return off;
  };
  const size_t headBytes = (size_t)kB * kH * kT * kDK * 2;  // 8 MB each
  size_t off_wq  = alloc((size_t)kInner * kInner * 2);
  size_t off_wk  = alloc((size_t)kInner * kInner * 2);
  size_t off_wv  = alloc((size_t)kInner * kInner * 2);
  size_t off_wo  = alloc((size_t)kInner * kInner * 2);
  size_t off_bo  = alloc((size_t)kInner * 4);
  size_t off_qu  = alloc(headBytes);
  size_t off_qv  = alloc(headBytes);
  size_t off_k   = alloc(headBytes);
  size_t off_vt  = alloc(headBytes);
  size_t off_p   = alloc((size_t)kH * kP * kDK * 2);
  size_t off_att = alloc((size_t)kB * kT * kInner * 2);  // total ~44 MB

  dim3 blk(128);
  const int Mrows = kB * kT;  // 8192

  // 1) fuse projection weights: Weff = Wqkv_slice @ w_{q,k,v};  Wout = w_ao @ w_o
  for (int i = 0; i < 3; ++i) {
    GemmArgs g{};
    g.Af = w_qkv + 512 * i; g.lda = 3 * kInner;
    g.Bf = (i == 0) ? w_q : (i == 1) ? w_k : w_v; g.ldb = kInner;
    g.M = kInner; g.N = kInner; g.K = kInner;
    g.out0 = (__bf16*)(ws + ((i == 0) ? off_wq : (i == 1) ? off_wk : off_wv));
    gemm_kernel<true, true, MODE_PLAIN><<<dim3(8, 8), blk, 0, stream>>>(g);
  }
  {
    GemmArgs g{};
    g.Af = w_ao; g.lda = kInner; g.Bf = w_o; g.ldb = kInner;
    g.M = kInner; g.N = kInner; g.K = kInner;
    g.out0 = (__bf16*)(ws + off_wo);
    gemm_kernel<true, true, MODE_PLAIN><<<dim3(8, 8), blk, 0, stream>>>(g);
  }
  bout_kernel<<<1, kInner, 0, stream>>>(b_ao, w_o, b_o, (float*)(ws + off_bo));

  // 2) projections (fp32 x -> bf16 head tensors)
  {
    GemmArgs g{};
    g.Af = x; g.lda = kInner;
    g.Bb = (const __bf16*)(ws + off_wq); g.ldb = kInner;
    g.M = Mrows; g.N = kInner; g.K = kInner;
    g.scale = 0.125f;  // 1/sqrt(DK)
    g.bias = b_q; g.bias_u = bias_u; g.bias_v = bias_v;
    g.out0 = (__bf16*)(ws + off_qu); g.out1 = (__bf16*)(ws + off_qv);
    gemm_kernel<true, false, MODE_QUV><<<dim3(8, Mrows / 64), blk, 0, stream>>>(g);
  }
  {
    GemmArgs g{};
    g.Af = x; g.lda = kInner;
    g.Bb = (const __bf16*)(ws + off_wk); g.ldb = kInner;
    g.M = Mrows; g.N = kInner; g.K = kInner;
    g.bias = b_k; g.out0 = (__bf16*)(ws + off_k);
    gemm_kernel<true, false, MODE_KP><<<dim3(8, Mrows / 64), blk, 0, stream>>>(g);
  }
  {
    GemmArgs g{};
    g.Af = x; g.lda = kInner;
    g.Bb = (const __bf16*)(ws + off_wv); g.ldb = kInner;
    g.M = Mrows; g.N = kInner; g.K = kInner;
    g.bias = b_v; g.out0 = (__bf16*)(ws + off_vt);
    gemm_kernel<true, false, MODE_VP><<<dim3(8, Mrows / 64), blk, 0, stream>>>(g);
  }
  {  // positional projection (M = 2047, row-clamped)
    GemmArgs g{};
    g.Af = pos; g.lda = kInner;
    g.Bf = w_pos; g.ldb = kInner;
    g.M = kP; g.N = kInner; g.K = kInner;
    g.out0 = (__bf16*)(ws + off_p);
    gemm_kernel<true, true, MODE_POS><<<dim3(8, (kP + 63) / 64), blk, 0, stream>>>(g);
  }

  // 3) flash attention with banded rel-shift, TDM-staged tiles
  attn_kernel<<<dim3(kT / 64, kB * kH), blk, 0, stream>>>(
      (const __bf16*)(ws + off_qu), (const __bf16*)(ws + off_qv),
      (const __bf16*)(ws + off_k),  (const __bf16*)(ws + off_vt),
      (const __bf16*)(ws + off_p),  (__bf16*)(ws + off_att));

  // 4) fused output projection: out = attn @ (w_ao@w_o) + b_out  (fp32 store)
  {
    GemmArgs g{};
    g.Ab = (const __bf16*)(ws + off_att); g.lda = kInner;
    g.Bb = (const __bf16*)(ws + off_wo);  g.ldb = kInner;
    g.M = Mrows; g.N = kInner; g.K = kInner;
    g.bias = (const float*)(ws + off_bo);
    g.outf = (float*)d_out;
    gemm_kernel<false, false, MODE_OUT><<<dim3(8, Mrows / 64), blk, 0, stream>>>(g);
  }
}
